// GroupedQueryAttention_73375221285372
// MI455X (gfx1250) — matrix operand order repaired and checked by reading, not compiled, not run on hardware
//
#include <hip/hip_runtime.h>

// ---------------------------------------------------------------------------
// GQA block for MI455X (gfx1250, wave32, WMMA).
// bf16 matrix paths, fp32 accumulation via v_wmma_f32_16x16x32_bf16.
// LDS panels are stored pre-swizzled into WMMA fragment order so each lane
// fetches its fragment with two ds_load_b128 (no VGPR repacking), with row
// pitches padded for conflict-free bank access.
// ---------------------------------------------------------------------------

typedef __attribute__((ext_vector_type(16))) __bf16 v16bf;
typedef __attribute__((ext_vector_type(8)))  float  v8f;

struct U4 { unsigned int x, y, z, w; };
union BF16Frag { v16bf v; unsigned int u[8]; };

#define GQA_B    2
#define GQA_S    2048
#define GQA_E    2048
#define GQA_H    16
#define GQA_HKV  4
#define GQA_D    128
#define GQA_EKV  (GQA_HKV * GQA_D)   // 512
#define GQA_M    (GQA_B * GQA_S)     // 4096

__device__ __forceinline__ unsigned short f32_to_bf16(float f) {
  unsigned int u = __float_as_uint(f);
  u += 0x7fffu + ((u >> 16) & 1u);     // round-to-nearest-even
  return (unsigned short)(u >> 16);
}

// group-swap swizzle on k-pair index: {0-3,4-7,8-11,12-15} -> {0-3,8-11,4-7,12-15}
__device__ __forceinline__ int kp_swz(int kp) {
  int g = kp >> 2;
  int gp = ((g & 1) << 1) | (g >> 1);
  return (gp << 2) | (kp & 3);
}

// ---------------------------------------------------------------------------
// Elementwise fp32 -> bf16
// ---------------------------------------------------------------------------
__global__ void cvt_f32_bf16(const float* __restrict__ X,
                             unsigned short* __restrict__ Y, int n) {
  int i = blockIdx.x * blockDim.x + threadIdx.x;
  if (i < n) Y[i] = f32_to_bf16(X[i]);
}

// ---------------------------------------------------------------------------
// RoPE on fp32 projection output, emit bf16 [B,S,heads,D]
// ---------------------------------------------------------------------------
__global__ void rope_f32_to_bf16(const float* __restrict__ X,
                                 unsigned short* __restrict__ Y,
                                 int heads, int total_half) {
  int idx = blockIdx.x * blockDim.x + threadIdx.x;
  if (idx >= total_half) return;
  const int half = GQA_D / 2;               // 64
  int d = idx % half;
  int t = idx / half;                        // (b*S + s)*heads + h
  int s = (t / heads) % GQA_S;
  size_t base = (size_t)t * GQA_D;
  float x1 = X[base + d];
  float x2 = X[base + d + half];
  float freq = __powf(10000.0f, -2.0f * (float)d / (float)GQA_D);
  float sn, cs;
  __sincosf((float)s * freq, &sn, &cs);
  Y[base + d]        = f32_to_bf16(x1 * cs - x2 * sn);
  Y[base + d + half] = f32_to_bf16(x2 * cs + x1 * sn);
}

// ---------------------------------------------------------------------------
// bf16 GEMM, fp32 out:  C[M,N] = A[M,K] * B[K,N]
// Block tile 64x64, K panel 32, 4 waves; wave = 16 rows x 64 cols.
// A panel LDS layout: A_sw[row][slot], slots = k-pairs permuted so that
//   lane-half hi reads its 8 fragment dwords at [hi*8 .. hi*8+7]  (2x b128).
// B panel LDS layout: B_sw[n][kp] (transposed), fragment = [hi*4..+3],[hi*4+8..+11] (2x b128).
// Row pitch 20 dwords: 16B-aligned rows, conflict-free lane->bank mapping.
// ---------------------------------------------------------------------------
#define APITCH 20
#define BPITCH 20

__global__ __launch_bounds__(128)
void gemm_bf16_f32(const unsigned short* __restrict__ A,
                   const unsigned short* __restrict__ Bm,
                   float* __restrict__ C, int M, int N, int K) {
  __shared__ __align__(16) unsigned int A_sw[64][APITCH];  // 5 KB
  __shared__ __align__(16) unsigned int B_sw[64][BPITCH];  // 5 KB

  const int tid  = threadIdx.x;
  const int lane = tid & 31;
  const int wave = tid >> 5;
  const int ln   = lane & 15;
  const int hi   = lane >> 4;
  const int bm   = blockIdx.y * 64;
  const int bn   = blockIdx.x * 64;

  v8f acc[4];
#pragma unroll
  for (int i = 0; i < 4; ++i)
#pragma unroll
    for (int j = 0; j < 8; ++j) acc[i][j] = 0.0f;

  for (int k0 = 0; k0 < K; k0 += 32) {
    // ---- stage A panel 64x32 bf16, fragment-order swizzled ----
    {
      int row  = tid >> 1;
      int cp   = (tid & 1) * 8;          // k-pair base 0 or 8
      int slot = cp >> 1;                // 0 or 4
      const U4* src = (const U4*)((const unsigned int*)
                        (A + (size_t)(bm + row) * K + k0) + cp);
      U4 t0 = src[0];                    // k-pairs cp+0..cp+3
      U4 t1 = src[1];                    // k-pairs cp+4..cp+7
      *(U4*)&A_sw[row][slot]     = t0;   // cp=0 -> slots 0-3 ; cp=8 -> 4-7
      *(U4*)&A_sw[row][slot + 8] = t1;   // cp=0 -> slots 8-11; cp=8 -> 12-15
      if (k0 + 32 < K)
        __builtin_prefetch((const void*)((const char*)src + 64), 0, 3);
    }
    // ---- stage B panel 32x64 bf16, transposed + pair-interleaved ----
    {
      int kp = tid & 15;
      int nb = (tid >> 4) * 8;
      const unsigned short* r0 = Bm + (size_t)(k0 + 2 * kp) * N + bn + nb;
      const unsigned short* r1 = r0 + N;
      U4 a0 = *(const U4*)r0;
      U4 b0 = *(const U4*)r1;
      unsigned int w0[4] = {a0.x, a0.y, a0.z, a0.w};
      unsigned int w1[4] = {b0.x, b0.y, b0.z, b0.w};
#pragma unroll
      for (int j = 0; j < 8; ++j) {
        unsigned int lo = (w0[j >> 1] >> ((j & 1) * 16)) & 0xffffu;
        unsigned int hh = (w1[j >> 1] >> ((j & 1) * 16)) & 0xffffu;
        B_sw[nb + j][kp] = lo | (hh << 16);
      }
      if (k0 + 32 < K)
        __builtin_prefetch((const void*)(r0 + 2 * (size_t)N * 16), 0, 3);
    }
    __syncthreads();

    // ---- fragments: 2x ds_load_b128 each ----
    BF16Frag af;
    {
      const U4* ap = (const U4*)&A_sw[wave * 16 + ln][hi * 8];
      *(U4*)&af.u[0] = ap[0];
      *(U4*)&af.u[4] = ap[1];
    }
#pragma unroll
    for (int nt = 0; nt < 4; ++nt) {
      BF16Frag bf;
      const U4* bp = (const U4*)&B_sw[nt * 16 + ln][hi * 4];
      *(U4*)&bf.u[0] = bp[0];
      *(U4*)&bf.u[4] = bp[2];
      acc[nt] = __builtin_amdgcn_wmma_f32_16x16x32_bf16(
          false, af.v, false, bf.v, (short)0, acc[nt], false, false);
    }
    __syncthreads();
  }

  // ---- store: C-frag lane n=L&15, vgpr r -> row r+8*hi ----
#pragma unroll
  for (int nt = 0; nt < 4; ++nt)
#pragma unroll
    for (int r = 0; r < 8; ++r) {
      int m = wave * 16 + r + 8 * hi;
      C[(size_t)(bm + m) * N + bn + nt * 16 + ln] = acc[nt][r];
    }
}

// ---------------------------------------------------------------------------
// Flash attention (non-causal, full softmax over S):
//   grid.x = b*H + h, grid.y = 64-row query block. 4 waves x 16 rows.
//   All 4 waves share one KV head (GQA) -> K/V tiles staged once in LDS.
// K tile pitch 68 dwords (64-pitch would put all lanes on one bank).
// ---------------------------------------------------------------------------
#define KPITCH 68
#define VPITCH 20
#define PPITCH 40   // ushorts

__global__ __launch_bounds__(128)
void flash_attn_bf16(const unsigned short* __restrict__ Qb,  // [B,S,H,D]
                     const unsigned short* __restrict__ Kb,  // [B,S,Hkv,D]
                     const unsigned short* __restrict__ Vb,  // [B,S,Hkv,D]
                     unsigned short* __restrict__ Ob) {      // [B,S,H,D]
  __shared__ __align__(16) unsigned int   Kl[32][KPITCH];        // ~8.5 KB
  __shared__ __align__(16) unsigned int   Vt[128][VPITCH];       // 10 KB
  __shared__ __align__(16) unsigned short Pb[4][16][PPITCH];     //  5 KB

  const int bh = blockIdx.x;
  const int b  = bh / GQA_H;
  const int h  = bh % GQA_H;
  const int hk = h / (GQA_H / GQA_HKV);
  const int q0 = blockIdx.y * 64;

  const int tid  = threadIdx.x;
  const int lane = tid & 31;
  const int wave = tid >> 5;
  const int ln   = lane & 15;
  const int hi   = lane >> 4;
  const float scale = 0.08838834764831845f;   // 1/sqrt(128)

  // ---- Q A-fragments (4 chunks of K=32 across D=128), loaded once ----
  BF16Frag qf[4];
  {
    int srow = q0 + wave * 16 + ln;
    const unsigned int* qrow = (const unsigned int*)
        (Qb + (((size_t)b * GQA_S + srow) * GQA_H + h) * GQA_D);
#pragma unroll
    for (int c = 0; c < 4; ++c)
#pragma unroll
      for (int i = 0; i < 8; ++i) {
        int kp = c * 16 + (i >> 2) * 8 + hi * 4 + (i & 3);
        qf[c].u[i] = qrow[kp];
      }
  }

  float rmax[8], rsum[8];
  v8f o[8];
#pragma unroll
  for (int r = 0; r < 8; ++r) { rmax[r] = -1e30f; rsum[r] = 0.0f; }
#pragma unroll
  for (int j = 0; j < 8; ++j)
#pragma unroll
    for (int r = 0; r < 8; ++r) o[j][r] = 0.0f;

  for (int kb = 0; kb < GQA_S; kb += 32) {
    // ---- stage K tile 32x128: [key][dpair], 4x b128 per thread ----
    {
      int key = tid >> 2;
      int dp  = (tid & 3) * 16;
      const U4* src = (const U4*)((const unsigned int*)
          (Kb + (((size_t)b * GQA_S + kb + key) * GQA_HKV + hk) * GQA_D) + dp);
      U4 x0 = src[0], x1 = src[1], x2 = src[2], x3 = src[3];
      U4* dst = (U4*)&Kl[key][dp];
      dst[0] = x0; dst[1] = x1; dst[2] = x2; dst[3] = x3;
    }
    // ---- stage V tile transposed + pair-interleaved along keys ----
    {
      int kp = tid & 15;
      int db = (tid >> 4) * 16;
      const unsigned short* r0 =
          Vb + (((size_t)b * GQA_S + kb + 2 * kp) * GQA_HKV + hk) * GQA_D + db;
      const unsigned short* r1 = r0 + GQA_HKV * GQA_D;
      const U4* p0 = (const U4*)r0;
      const U4* p1 = (const U4*)r1;
      U4 a0 = p0[0], a1 = p0[1], c0 = p1[0], c1 = p1[1];
      unsigned int w0[8] = {a0.x, a0.y, a0.z, a0.w, a1.x, a1.y, a1.z, a1.w};
      unsigned int w1[8] = {c0.x, c0.y, c0.z, c0.w, c1.x, c1.y, c1.z, c1.w};
#pragma unroll
      for (int j = 0; j < 16; ++j) {
        unsigned int lo = (w0[j >> 1] >> ((j & 1) * 16)) & 0xffffu;
        unsigned int hh = (w1[j >> 1] >> ((j & 1) * 16)) & 0xffffu;
        Vt[db + j][kp] = lo | (hh << 16);
      }
    }
    __syncthreads();

    // ---- scores: S = Q * K^T (two 16x16 frags over 32 keys) ----
    v8f s0, s1;
#pragma unroll
    for (int r = 0; r < 8; ++r) { s0[r] = 0.0f; s1[r] = 0.0f; }
#pragma unroll
    for (int c = 0; c < 4; ++c) {
      BF16Frag kf0, kf1;
      {
        const U4* k0p = (const U4*)&Kl[ln][c * 16 + hi * 4];
        *(U4*)&kf0.u[0] = k0p[0];
        *(U4*)&kf0.u[4] = k0p[2];
        const U4* k1p = (const U4*)&Kl[16 + ln][c * 16 + hi * 4];
        *(U4*)&kf1.u[0] = k1p[0];
        *(U4*)&kf1.u[4] = k1p[2];
      }
      s0 = __builtin_amdgcn_wmma_f32_16x16x32_bf16(
          false, qf[c].v, false, kf0.v, (short)0, s0, false, false);
      s1 = __builtin_amdgcn_wmma_f32_16x16x32_bf16(
          false, qf[c].v, false, kf1.v, (short)0, s1, false, false);
    }

    // ---- online softmax (rows live in 16-lane halves) ----
#pragma unroll
    for (int r = 0; r < 8; ++r) {
      float a  = s0[r] * scale;
      float bb = s1[r] * scale;
      float mx = fmaxf(a, bb);
#pragma unroll
      for (int m = 1; m < 16; m <<= 1) mx = fmaxf(mx, __shfl_xor(mx, m, 32));
      float nm = fmaxf(rmax[r], mx);
      float corr = __expf(rmax[r] - nm);
      rmax[r] = nm;
      float p0 = __expf(a - nm);
      float p1 = __expf(bb - nm);
      s0[r] = p0; s1[r] = p1;
      float ts = p0 + p1;
#pragma unroll
      for (int m = 1; m < 16; m <<= 1) ts += __shfl_xor(ts, m, 32);
      rsum[r] = rsum[r] * corr + ts;
#pragma unroll
      for (int j = 0; j < 8; ++j) o[j][r] *= corr;
    }

    // ---- P (C-layout, f32) -> LDS (swizzled) -> A-fragment via 2x b128 ----
    {
      int kpa = kp_swz(ln >> 1);        // key ln
      int kpb = kp_swz(8 + (ln >> 1));  // key 16+ln
      int par = ln & 1;
#pragma unroll
      for (int r = 0; r < 8; ++r) {
        int row = r + 8 * hi;
        Pb[wave][row][2 * kpa + par] = f32_to_bf16(s0[r]);
        Pb[wave][row][2 * kpb + par] = f32_to_bf16(s1[r]);
      }
    }
    __syncthreads();

    BF16Frag pf;
    {
      const U4* pp = (const U4*)((const unsigned int*)&Pb[wave][ln][0] + hi * 8);
      *(U4*)&pf.u[0] = pp[0];
      *(U4*)&pf.u[4] = pp[1];
    }
    // ---- O += P * V (8 d-tiles of 16 cols), V frags 2x b128 each ----
#pragma unroll
    for (int j = 0; j < 8; ++j) {
      BF16Frag vf;
      const U4* vp = (const U4*)&Vt[j * 16 + ln][hi * 4];
      *(U4*)&vf.u[0] = vp[0];
      *(U4*)&vf.u[4] = vp[2];
      o[j] = __builtin_amdgcn_wmma_f32_16x16x32_bf16(
          false, pf.v, false, vf.v, (short)0, o[j], false, false);
    }
    __syncthreads();
  }

  // ---- normalize and store bf16 attention output ----
  float inv[8];
#pragma unroll
  for (int r = 0; r < 8; ++r) inv[r] = 1.0f / rsum[r];
  int sbase = q0 + wave * 16;
#pragma unroll
  for (int j = 0; j < 8; ++j)
#pragma unroll
    for (int r = 0; r < 8; ++r) {
      int m = r + 8 * hi;
      int d = j * 16 + ln;
      Ob[(((size_t)b * GQA_S + sbase + m) * GQA_H + h) * GQA_D + d] =
          f32_to_bf16(o[j][r] * inv[r]);
    }
}

// ---------------------------------------------------------------------------
// Host-side orchestration
// ---------------------------------------------------------------------------
extern "C" void kernel_launch(void* const* d_in, const int* in_sizes, int n_in,
                              void* d_out, int out_size, void* d_ws, size_t ws_size,
                              hipStream_t stream) {
  (void)in_sizes; (void)n_in; (void)out_size; (void)ws_size;
  const float* hs = (const float*)d_in[0];
  const float* wq = (const float*)d_in[1];
  const float* wk = (const float*)d_in[2];
  const float* wv = (const float*)d_in[3];
  const float* wo = (const float*)d_in[4];
  float* out = (float*)d_out;

  const int M = GQA_M, E = GQA_E, EKV = GQA_EKV;

  char* w = (char*)d_ws;
  size_t off = 0;
  auto take = [&](size_t bytes) -> void* {
    void* p = (void*)(w + off);
    off = (off + bytes + 255) & ~(size_t)255;
    return p;
  };
  unsigned short* hsb = (unsigned short*)take((size_t)M * E * 2);
  unsigned short* wqb = (unsigned short*)take((size_t)E * E * 2);
  unsigned short* wkb = (unsigned short*)take((size_t)E * EKV * 2);
  unsigned short* wvb = (unsigned short*)take((size_t)E * EKV * 2);
  unsigned short* wob = (unsigned short*)take((size_t)E * E * 2);
  float*          qf  = (float*)take((size_t)M * E * 4);
  float*          kf  = (float*)take((size_t)M * EKV * 4);
  float*          vf  = (float*)take((size_t)M * EKV * 4);
  unsigned short* qb  = (unsigned short*)take((size_t)M * E * 2);
  unsigned short* kb  = (unsigned short*)take((size_t)M * EKV * 2);
  unsigned short* vb  = (unsigned short*)take((size_t)M * EKV * 2);
  unsigned short* ab  = (unsigned short*)take((size_t)M * E * 2);

  // 1) fp32 -> bf16 for activations + weights
  {
    int n;
    n = M * E;   cvt_f32_bf16<<<(n + 255) / 256, 256, 0, stream>>>(hs, hsb, n);
    n = E * E;   cvt_f32_bf16<<<(n + 255) / 256, 256, 0, stream>>>(wq, wqb, n);
    n = E * EKV; cvt_f32_bf16<<<(n + 255) / 256, 256, 0, stream>>>(wk, wkb, n);
    n = E * EKV; cvt_f32_bf16<<<(n + 255) / 256, 256, 0, stream>>>(wv, wvb, n);
    n = E * E;   cvt_f32_bf16<<<(n + 255) / 256, 256, 0, stream>>>(wo, wob, n);
  }

  // 2) Q/K/V projections (WMMA bf16 GEMM, fp32 out)
  gemm_bf16_f32<<<dim3(E / 64,   M / 64), 128, 0, stream>>>(hsb, wqb, qf, M, E,   E);
  gemm_bf16_f32<<<dim3(EKV / 64, M / 64), 128, 0, stream>>>(hsb, wkb, kf, M, EKV, E);
  gemm_bf16_f32<<<dim3(EKV / 64, M / 64), 128, 0, stream>>>(hsb, wvb, vf, M, EKV, E);

  // 3) RoPE -> bf16 (Q with 16 heads, K with 4 heads), V plain convert
  {
    int nq = M * GQA_H * (GQA_D / 2);
    rope_f32_to_bf16<<<(nq + 255) / 256, 256, 0, stream>>>(qf, qb, GQA_H, nq);
    int nk = M * GQA_HKV * (GQA_D / 2);
    rope_f32_to_bf16<<<(nk + 255) / 256, 256, 0, stream>>>(kf, kb, GQA_HKV, nk);
    int nv = M * EKV;
    cvt_f32_bf16<<<(nv + 255) / 256, 256, 0, stream>>>(vf, vb, nv);
  }

  // 4) Flash attention (GQA), bf16 out [B,S,H,D] == [M,E]
  flash_attn_bf16<<<dim3(GQA_B * GQA_H, GQA_S / 64), 128, 0, stream>>>(qb, kb, vb, ab);

  // 5) Output projection -> fp32 d_out
  gemm_bf16_f32<<<dim3(E / 64, M / 64), 128, 0, stream>>>(ab, wob, out, M, E, E);
}
